// BloomSelfAttention_10685878633227
// MI455X (gfx1250) — compile-verified, matrix operand order checked by reading, not run on hardware
//
#include <hip/hip_runtime.h>
#include <hip/hip_bf16.h>
#include <stdint.h>

#define NHEAD 16
#define HD    128
#define SEQ   2048
#define DIMK  2048
#define N3    6144
#define NBATCH 4
#define NEG_BIG (-1e10f)

typedef __attribute__((ext_vector_type(16))) __bf16 v16bf;
typedef __attribute__((ext_vector_type(8)))  float  v8f;

union FragBF {
    v16bf v;
    uint4 q[2];
    __bf16 e[16];
};

static __device__ __forceinline__ __bf16 f2bf(float f) {
    union { float f; uint32_t u; } a; a.f = f;
    uint32_t r = a.u + 0x7FFFu + ((a.u >> 16) & 1u);
    union { uint16_t s; __bf16 b; } o; o.s = (uint16_t)(r >> 16);
    return o.b;
}

// ---------------------------------------------------------------------------
// Kernel 1: fused QKV projection.  C[8192,6144] = A[8192,2048] x W[2048,6144]+b
// bf16 WMMA with fp32 accumulate; epilogue de-interleaves (d*3+{q,v,k}) columns
// and stores bf16 Q (pre-scaled by 1/sqrt(hd)), V, K into workspace as
// [b][h][s][hd] so the attention pass streams them from L2.
// ---------------------------------------------------------------------------
__global__ __launch_bounds__(256)
void qkv_gemm_kernel(const float* __restrict__ A, const float* __restrict__ W,
                     const float* __restrict__ bias,
                     __bf16* __restrict__ Qb, __bf16* __restrict__ Vb,
                     __bf16* __restrict__ Kb) {
    __shared__ __bf16 Asm[128][48];   // 128 rows x 32 k (stride 48 -> 96B rows)
    __shared__ __bf16 Bsm[128][48];   // transposed: [n][k]

    const int tid  = threadIdx.x;
    const int lane = tid & 31;
    const int wave = tid >> 5;
    const int wm   = wave & 3;            // 4 waves over M, 32 rows each
    const int wn   = wave >> 2;           // 2 waves over N, 64 cols each
    const int m0   = blockIdx.y * 128;
    const int n0   = blockIdx.x * 128;

    const int lr   = lane & 15;
    const int half = lane >> 4;
    const int kb   = half * 8;            // ISA A/B frag K-group base

    const v8f vzero = {0.f,0.f,0.f,0.f,0.f,0.f,0.f,0.f};
    v8f acc[2][4];
#pragma unroll
    for (int i = 0; i < 2; ++i)
#pragma unroll
        for (int j = 0; j < 4; ++j) acc[i][j] = vzero;

    const int arow = tid >> 3;            // 0..31
    const int acol = (tid & 7) * 4;       // 0..28
    const int brow = tid >> 5;            // 0..7
    const int bcol = (tid & 31) * 4;      // 0..124

    for (int k0 = 0; k0 < DIMK; k0 += 32) {
        // stage A tile (fp32 -> bf16), coalesced float4 reads
#pragma unroll
        for (int p = 0; p < 4; ++p) {
            int r = arow + p * 32;
            float4 f = *(const float4*)(A + (size_t)(m0 + r) * DIMK + k0 + acol);
            union { uint2 u; __bf16 e[4]; } pk;
            pk.e[0] = f2bf(f.x); pk.e[1] = f2bf(f.y);
            pk.e[2] = f2bf(f.z); pk.e[3] = f2bf(f.w);
            *(uint2*)&Asm[r][acol] = pk.u;
        }
        // stage B tile transposed into [n][k]
#pragma unroll
        for (int p = 0; p < 4; ++p) {
            int kk = brow + p * 8;
            float4 f = *(const float4*)(W + (size_t)(k0 + kk) * N3 + n0 + bcol);
            Bsm[bcol + 0][kk] = f2bf(f.x);
            Bsm[bcol + 1][kk] = f2bf(f.y);
            Bsm[bcol + 2][kk] = f2bf(f.z);
            Bsm[bcol + 3][kk] = f2bf(f.w);
        }
        __syncthreads();

        FragBF af[2], bfb[4];
#pragma unroll
        for (int mt = 0; mt < 2; ++mt) {
            const __bf16* p = &Asm[wm * 32 + mt * 16 + lr][kb];
            af[mt].q[0] = *(const uint4*)p;
            af[mt].q[1] = *(const uint4*)(p + 16);
        }
#pragma unroll
        for (int nt = 0; nt < 4; ++nt) {
            const __bf16* p = &Bsm[wn * 64 + nt * 16 + lr][kb];
            bfb[nt].q[0] = *(const uint4*)p;
            bfb[nt].q[1] = *(const uint4*)(p + 16);
        }
#pragma unroll
        for (int mt = 0; mt < 2; ++mt)
#pragma unroll
            for (int nt = 0; nt < 4; ++nt)
                acc[mt][nt] = __builtin_amdgcn_wmma_f32_16x16x32_bf16(
                    false, af[mt].v, false, bfb[nt].v, (short)0, acc[mt][nt],
                    false, false);
        __syncthreads();
    }

    // epilogue: bias add, split q/v/k (col e -> type e%3, feature e/3), store bf16
    const float qscale = 0.08838834764831845f;  // 1/sqrt(128)
#pragma unroll
    for (int mt = 0; mt < 2; ++mt) {
#pragma unroll
        for (int nt = 0; nt < 4; ++nt) {
            int e  = n0 + wn * 64 + nt * 16 + lr;
            float bv = bias[e];
            int t  = e % 3;
            int d  = e / 3;
            int hh = d >> 7;
            int dm = d & 127;
#pragma unroll
            for (int r = 0; r < 8; ++r) {
                int rowg = m0 + wm * 32 + mt * 16 + r + half * 8;
                int bb = rowg >> 11;
                int s  = rowg & 2047;
                size_t idx = ((size_t)((bb * NHEAD + hh) * SEQ + s)) * HD + dm;
                float v = acc[mt][nt][r] + bv;
                if (t == 0)      Qb[idx] = f2bf(v * qscale);
                else if (t == 1) Vb[idx] = f2bf(v);
                else             Kb[idx] = f2bf(v);
            }
        }
    }
}

// ---------------------------------------------------------------------------
// Kernel 2: causal flash attention.  Each wave owns 16 query rows; the
// workgroup (8 waves = 128 rows) cooperatively stages 32-key K/V blocks in
// LDS.  K staging uses GLOBAL_LOAD_ASYNC_TO_LDS_B128 (ASYNCcnt path) since
// the data is already bf16; V needs a transpose so it goes through VGPRs.
// Q.K^T and P.V both run on v_wmma_f32_16x16x32_bf16; softmax is fp32 with
// per-row wave32 shuffle reductions.
// ---------------------------------------------------------------------------
__global__ __launch_bounds__(256)
void attn_kernel(const __bf16* __restrict__ Qb, const __bf16* __restrict__ Kb,
                 const __bf16* __restrict__ Vb, float* __restrict__ out) {
    __shared__ __bf16 Ksm[32][136];      // [key][hd]    row stride 272B
    __shared__ __bf16 Vsm[128][48];      // [hd][key]    row stride 96B
    __shared__ __bf16 Psm[8][16][48];    // per-wave P transpose patch

    const int tid  = threadIdx.x;
    const int lane = tid & 31;
    const int wave = tid >> 5;
    const int q0   = blockIdx.x * 128;
    const int h    = blockIdx.y;
    const int b    = blockIdx.z;
    const int qw   = q0 + wave * 16;

    const int lr   = lane & 15;
    const int half = lane >> 4;
    const int kb   = half * 8;

    const size_t base = ((size_t)(b * NHEAD + h)) * SEQ * HD;
    const __bf16* Qh = Qb + base;
    const __bf16* Kh = Kb + base;
    const __bf16* Vh = Vb + base;

    // Q fragments: 16x128 = 4 A-frags (K=32 each), straight from global bf16
    FragBF qf[4];
#pragma unroll
    for (int kf = 0; kf < 4; ++kf) {
        const __bf16* p = Qh + (size_t)(qw + lr) * HD + kf * 32 + kb;
        qf[kf].q[0] = *(const uint4*)p;
        qf[kf].q[1] = *(const uint4*)(p + 16);
    }

    const v8f vzero = {0.f,0.f,0.f,0.f,0.f,0.f,0.f,0.f};
    v8f O[8];
    float m[8], l[8];
#pragma unroll
    for (int i = 0; i < 8; ++i) { O[i] = vzero; m[i] = -__builtin_inff(); l[i] = 0.f; }

    const int nblk = q0 / 32 + 4;        // causal: blocks covering keys <= q0+127
    const int skey = tid >> 3;           // 0..31  staging key row
    const int seg  = tid & 7;            // 0..7   16-element hd segment

    // async staging constants: GVS mode (SGPR64 base + per-lane VGPR offset)
    const uint64_t kbase64 = (uint64_t)(uintptr_t)Kh;
    // LDS destination byte address for this thread's 32B K slice (addr[31:0])
    const uint32_t klds = (uint32_t)(uintptr_t)&Ksm[skey][seg * 16];

    for (int j = 0; j < nblk; ++j) {
        const int jb = j * 32;
        // ---- stage K block via async global->LDS DMA (no VGPR round-trip) ----
        {
            const uint32_t goff =
                (uint32_t)(((jb + skey) * HD + seg * 16) * sizeof(__bf16));
            asm volatile(
                "global_load_async_to_lds_b128 %0, %1, %2\n\t"
                "global_load_async_to_lds_b128 %0, %1, %2 offset:16"
                :
                : "v"(klds), "v"(goff), "s"(kbase64)
                : "memory");
        }
        // ---- stage V block transposed through VGPRs ----
        {
            union { uint4 q[2]; __bf16 e[16]; } vv;
            const uint4* vs = (const uint4*)(Vh + (size_t)(jb + skey) * HD) + seg * 2;
            vv.q[0] = vs[0]; vv.q[1] = vs[1];
#pragma unroll
            for (int i = 0; i < 16; ++i)
                Vsm[seg * 16 + i][skey] = vv.e[i];

            if (j + 1 < nblk)   // global_prefetch_b8 of next V block
                __builtin_prefetch(Vh + (size_t)(jb + 32 + skey) * HD + seg * 16, 0, 3);
        }
        // producer waits for its async DMA, then workgroup barrier publishes LDS
        asm volatile("s_wait_asynccnt 0x0" ::: "memory");
        __syncthreads();

        // ---- scores: 2 key subtiles x 4 k-steps = 8 WMMA ----
        v8f sc[2];
#pragma unroll
        for (int st = 0; st < 2; ++st) {
            v8f a = vzero;
#pragma unroll
            for (int kf = 0; kf < 4; ++kf) {
                FragBF kfr;
                const __bf16* p = &Ksm[st * 16 + lr][kf * 32 + kb];
                kfr.q[0] = *(const uint4*)p;
                kfr.q[1] = *(const uint4*)(p + 16);
                a = __builtin_amdgcn_wmma_f32_16x16x32_bf16(
                        false, qf[kf].v, false, kfr.v, (short)0, a, false, false);
            }
            sc[st] = a;
        }

        // ---- causal mask + online softmax (per-row over 32 keys) ----
        float alpha[8], p0[8], p1[8];
#pragma unroll
        for (int r = 0; r < 8; ++r) {
            int rowg = qw + r + half * 8;
            float s0 = sc[0][r], s1 = sc[1][r];
            if (jb + lr > rowg)      s0 = NEG_BIG;
            if (jb + 16 + lr > rowg) s1 = NEG_BIG;
            float mx = fmaxf(s0, s1);
#pragma unroll
            for (int msk = 1; msk <= 8; msk <<= 1)
                mx = fmaxf(mx, __shfl_xor(mx, msk, 32));
            float mn = fmaxf(m[r], mx);
            float al = __expf(m[r] - mn);
            float e0 = __expf(s0 - mn);
            float e1 = __expf(s1 - mn);
            float rs = e0 + e1;
#pragma unroll
            for (int msk = 1; msk <= 8; msk <<= 1)
                rs += __shfl_xor(rs, msk, 32);
            l[r] = l[r] * al + rs;
            m[r] = mn;
            alpha[r] = al;
            p0[r] = e0;
            p1[r] = e1;
        }

        // rescale running output
#pragma unroll
        for (int t = 0; t < 8; ++t)
#pragma unroll
            for (int r = 0; r < 8; ++r)
                O[t][r] *= alpha[r];

        // ---- P: C-layout -> A-fragment via per-wave LDS patch ----
#pragma unroll
        for (int r = 0; r < 8; ++r) {
            Psm[wave][r + half * 8][lr]      = f2bf(p0[r]);
            Psm[wave][r + half * 8][16 + lr] = f2bf(p1[r]);
        }
        FragBF pf;
        {
            const __bf16* p = &Psm[wave][lr][kb];   // same-wave DS ops are in-order
            pf.q[0] = *(const uint4*)p;
            pf.q[1] = *(const uint4*)(p + 16);
        }

        // ---- O += P.V : 8 WMMA across hd ----
#pragma unroll
        for (int nt = 0; nt < 8; ++nt) {
            FragBF vf;
            const __bf16* p = &Vsm[nt * 16 + lr][kb];
            vf.q[0] = *(const uint4*)p;
            vf.q[1] = *(const uint4*)(p + 16);
            O[nt] = __builtin_amdgcn_wmma_f32_16x16x32_bf16(
                        false, pf.v, false, vf.v, (short)0, O[nt], false, false);
        }
        __syncthreads();
    }

    // ---- normalize and store fp32 output (B,S,H,hd) flattened ----
#pragma unroll
    for (int nt = 0; nt < 8; ++nt) {
        int dm = nt * 16 + lr;
#pragma unroll
        for (int r = 0; r < 8; ++r) {
            int s = qw + r + half * 8;
            float inv = 1.0f / l[r];
            out[((size_t)(b * SEQ + s) * NHEAD + h) * HD + dm] = O[nt][r] * inv;
        }
    }
}

// ---------------------------------------------------------------------------
extern "C" void kernel_launch(void* const* d_in, const int* in_sizes, int n_in,
                              void* d_out, int out_size, void* d_ws, size_t ws_size,
                              hipStream_t stream) {
    const float* hs   = (const float*)d_in[0];   // (4,2048,2048) fp32
    const float* W    = (const float*)d_in[1];   // (2048,6144)   fp32
    const float* bias = (const float*)d_in[2];   // (6144,)       fp32

    const size_t qkv_elems = (size_t)NBATCH * NHEAD * SEQ * HD;  // 16.7M each
    __bf16* Qb = (__bf16*)d_ws;
    __bf16* Vb = Qb + qkv_elems;
    __bf16* Kb = Vb + qkv_elems;   // 96 MB total -> fits in 192 MB L2

    dim3 g1(N3 / 128, (NBATCH * SEQ) / 128);     // 48 x 64 tiles
    qkv_gemm_kernel<<<g1, 256, 0, stream>>>(hs, W, bias, Qb, Vb, Kb);

    dim3 g2(SEQ / 128, NHEAD, NBATCH);           // 16 x 16 x 4 workgroups
    attn_kernel<<<g2, 256, 0, stream>>>(Qb, Kb, Vb, (float*)d_out);
}
